// ContextPredictionModel_54099408060660
// MI455X (gfx1250) — matrix operand order repaired and checked by reading, not compiled, not run on hardware
//
#include <hip/hip_runtime.h>
#include <hip/hip_bf16.h>

typedef __attribute__((ext_vector_type(16))) _Float16 v16h;
typedef __attribute__((ext_vector_type(8)))  _Float16 v8h;
typedef __attribute__((ext_vector_type(8)))  float    v8f;

#define C_DIM   1024
#define B_DIM   256
#define NPATCH  25
#define MROWS   (B_DIM * 9)   // 2304
#define EPSV    1e-5f
#define LDSTRIDE 40           // 32 + 8 halves pad -> 80B rows (16B aligned, bank-staggered)

struct PredTab { unsigned short e[120]; }; // (p<<8) | (d*3+s)

__device__ __forceinline__ v16h join8(v8h lo, v8h hi) {
  v16h r;
  #pragma unroll
  for (int u = 0; u < 8; ++u) { r[u] = lo[u]; r[u + 8] = hi[u]; }
  return r;
}

// ---------------------------------------------------------------------------
// Tiled f16 WMMA GEMM body: Y[128 x 1024-slab] = A[128 x 1024] * W^T + bias
//   A : fp16 row-major [m][k] slab (this block's 128 rows)
//   W : fp16 row-major [n][k]  (B-matrix: each row n is a column of B, K-major)
// Block: 256 threads = 8 waves (2 along M x 4 along N), each wave 4x2 16x16 tiles
// ---------------------------------------------------------------------------
template<bool F32OUT>
__device__ __forceinline__ void gemm_body(
    const _Float16* __restrict__ Ab,
    const _Float16* __restrict__ Wb,
    const float*    __restrict__ bias,
    float*          __restrict__ outF,
    _Float16*       __restrict__ outH,
    int n0)
{
  __shared__ __align__(16) _Float16 lA[2][128][LDSTRIDE];
  __shared__ __align__(16) _Float16 lB[2][128][LDSTRIDE];

  const int tid  = threadIdx.x;
  const int lane = tid & 31;
  const int wave = tid >> 5;
  const int wm   = wave & 1;        // 2 waves along M (64 rows each)
  const int wn   = wave >> 1;       // 4 waves along N (32 cols each)
  const int lrow = tid >> 1;        // tile row this thread stages (0..127)
  const int lseg = (tid & 1) * 16;  // first/second 16 halves of the K=32 strip

  const _Float16* gA = Ab + (size_t)lrow * C_DIM + lseg;
  const _Float16* gB = Wb + (size_t)(n0 + lrow) * C_DIM + lseg;

  const v8f zacc = {};
  v8f acc[4][2];
  #pragma unroll
  for (int i = 0; i < 4; ++i)
    #pragma unroll
    for (int j = 0; j < 2; ++j) acc[i][j] = zacc;

  // stage K-tile 0
  v8h ra0 = *(const v8h*)(gA + 0);
  v8h ra1 = *(const v8h*)(gA + 8);
  v8h rb0 = *(const v8h*)(gB + 0);
  v8h rb1 = *(const v8h*)(gB + 8);
  *(v8h*)&lA[0][lrow][lseg]     = ra0;
  *(v8h*)&lA[0][lrow][lseg + 8] = ra1;
  *(v8h*)&lB[0][lrow][lseg]     = rb0;
  *(v8h*)&lB[0][lrow][lseg + 8] = rb1;
  __syncthreads();

  // CDNA5 f16 fragment addressing (ISA 7.12.2):
  //  A 16x32: lane<16 -> K 0..7 & 16..23 of row (lane&15); lane>=16 -> K 8..15 & 24..31
  //  B 32x16: lane<16 -> K 0..7 & 16..23 of column (lane&15); lane>=16 -> K 8..15 & 24..31
  const int aoff = (lane < 16) ? 0 : 8;
  const int boff = (lane < 16) ? 0 : 8;
  const int am   = wm * 64 + (lane & 15);
  const int bn   = wn * 32 + (lane & 15);

  const int KT = C_DIM / 32;
  for (int kt = 0; kt < KT; ++kt) {
    const int cur = kt & 1;
    if (kt + 1 < KT) {
      const _Float16* pa = gA + (size_t)(kt + 1) * 32;
      const _Float16* pb = gB + (size_t)(kt + 1) * 32;
      ra0 = *(const v8h*)(pa + 0);
      ra1 = *(const v8h*)(pa + 8);
      rb0 = *(const v8h*)(pb + 0);
      rb1 = *(const v8h*)(pb + 8);
      if (kt + 3 < KT) {                       // global_prefetch_b8 next-next tile
        __builtin_prefetch(gA + (size_t)(kt + 3) * 32, 0, 0);
        __builtin_prefetch(gB + (size_t)(kt + 3) * 32, 0, 0);
      }
    }

    v16h a[4], b[2];
    #pragma unroll
    for (int i = 0; i < 4; ++i) {
      v8h lo = *(const v8h*)&lA[cur][am + i * 16][aoff];
      v8h hi = *(const v8h*)&lA[cur][am + i * 16][aoff + 16];
      a[i] = join8(lo, hi);
    }
    #pragma unroll
    for (int j = 0; j < 2; ++j) {
      v8h lo = *(const v8h*)&lB[cur][bn + j * 16][boff];
      v8h hi = *(const v8h*)&lB[cur][bn + j * 16][boff + 16];
      b[j] = join8(lo, hi);
    }

    #pragma unroll
    for (int i = 0; i < 4; ++i)
      #pragma unroll
      for (int j = 0; j < 2; ++j)
        acc[i][j] = __builtin_amdgcn_wmma_f32_16x16x32_f16(
            false, a[i], false, b[j], (short)0, acc[i][j], false, false);

    if (kt + 1 < KT) {
      const int nxt = (kt + 1) & 1;
      *(v8h*)&lA[nxt][lrow][lseg]     = ra0;
      *(v8h*)&lA[nxt][lrow][lseg + 8] = ra1;
      *(v8h*)&lB[nxt][lrow][lseg]     = rb0;
      *(v8h*)&lB[nxt][lrow][lseg + 8] = rb1;
      __syncthreads();
    }
  }

  // C/D layout: VGPR r -> row r (lanes 0-15) / row 8+r (lanes 16-31), col = lane&15
  const int hi8 = (lane >> 4) * 8;
  const int nc  = lane & 15;
  #pragma unroll
  for (int j = 0; j < 2; ++j) {
    const int n = n0 + wn * 32 + j * 16 + nc;
    const float bv = bias[n];
    #pragma unroll
    for (int i = 0; i < 4; ++i) {
      const int mbase = wm * 64 + i * 16 + hi8;
      #pragma unroll
      for (int r = 0; r < 8; ++r) {
        const float v = acc[i][j][r] + bv;
        const size_t idx = (size_t)(mbase + r) * C_DIM + n;
        if constexpr (F32OUT) outF[idx] = v;
        else                  outH[idx] = (_Float16)v;
      }
    }
  }
}

__global__ __launch_bounds__(256)
void gemm_conv_kernel(const _Float16* __restrict__ A, const _Float16* __restrict__ W,
                      const float* __restrict__ bias, _Float16* __restrict__ Y, int M)
{
  const size_t slab = ((size_t)blockIdx.z * M + (size_t)blockIdx.y * 128) * C_DIM;
  gemm_body<false>(A + slab, W, bias, nullptr, Y + slab, blockIdx.x * 128);
}

__global__ __launch_bounds__(256)
void gemm_head_kernel(const _Float16* __restrict__ ctx, const _Float16* __restrict__ Wl,
                      const float* __restrict__ lb, float* __restrict__ out, PredTab tab)
{
  const int q = blockIdx.z;
  const int e = tab.e[q];
  const int p = e >> 8, h = e & 255;
  const size_t arow = ((size_t)p * B_DIM + (size_t)blockIdx.y * 128) * C_DIM;
  const size_t orow = ((size_t)q * B_DIM + (size_t)blockIdx.y * 128) * C_DIM;
  gemm_body<true>(ctx + arow, Wl + (size_t)h * C_DIM * C_DIM, lb + (size_t)h * C_DIM,
                  out + orow, nullptr, blockIdx.x * 128);
}

// ---------------------------------------------------------------------------
// Support kernels
// ---------------------------------------------------------------------------
__global__ __launch_bounds__(256)
void f32_to_f16_kernel(const float* __restrict__ src, _Float16* __restrict__ dst, int n) {
  int i = blockIdx.x * 256 + threadIdx.x;
  if (i < n) dst[i] = (_Float16)src[i];
}

__global__ __launch_bounds__(256)
void pack_patches_kernel(const float* __restrict__ x, _Float16* __restrict__ H) {
  long long i = (long long)blockIdx.x * 256 + threadIdx.x;
  if (i >= (long long)NPATCH * MROWS * C_DIM) return;
  int c = (int)(i & (C_DIM - 1));
  long long t = i >> 10;                 // p*2304 + m
  int m = (int)(t % MROWS);
  int p = (int)(t / MROWS);
  int b = m / 9, s = m - b * 9;
  int y = s / 3, xx = s - y * 3;
  int py = p / 5, px = p - py * 5;
  float v = x[(((size_t)b * C_DIM + c) * 7 + (py + y)) * 7 + (px + xx)];
  H[i] = (_Float16)v;
}

__global__ __launch_bounds__(256)
void zero_f32_kernel(float* p, int n) {
  int i = blockIdx.x * 256 + threadIdx.x;
  if (i < n) p[i] = 0.f;
}

__global__ __launch_bounds__(256)
void bn_partial_kernel(const _Float16* __restrict__ H,
                       float* __restrict__ psum, float* __restrict__ psq) {
  const int p = blockIdx.x, mc = blockIdx.y;
  const int tid = threadIdx.x;
  const _Float16* base = H + ((size_t)p * MROWS + (size_t)mc * 128) * C_DIM;
  #pragma unroll
  for (int cc = 0; cc < 4; ++cc) {
    const int c = cc * 256 + tid;
    float s = 0.f, qq = 0.f;
    for (int r = 0; r < 128; ++r) {
      float v = (float)base[(size_t)r * C_DIM + c];
      s += v; qq += v * v;
    }
    atomicAdd(&psum[p * C_DIM + c], s);
    atomicAdd(&psq [p * C_DIM + c], qq);
  }
}

__global__ __launch_bounds__(256)
void bn_final_kernel(const float* __restrict__ psum, const float* __restrict__ psq,
                     const float* __restrict__ gamma, const float* __restrict__ beta,
                     float* __restrict__ scale, float* __restrict__ shift) {
  int i = blockIdx.x * 256 + threadIdx.x;          // p*1024 + c
  if (i >= NPATCH * C_DIM) return;
  int c = i & (C_DIM - 1);
  float mean = psum[i] * (1.f / MROWS);
  float var  = psq[i]  * (1.f / MROWS) - mean * mean;
  float rstd = rsqrtf(var + EPSV);
  float sc   = gamma[c] * rstd;
  scale[i] = sc;
  shift[i] = beta[c] - mean * sc;
}

__global__ __launch_bounds__(256)
void bn_apply_kernel(_Float16* __restrict__ H,
                     const float* __restrict__ scale, const float* __restrict__ shift) {
  long long i = (long long)blockIdx.x * 256 + threadIdx.x;
  if (i >= (long long)NPATCH * MROWS * C_DIM) return;
  int c = (int)(i & (C_DIM - 1));
  int p = (int)(i / ((long long)MROWS * C_DIM));
  float v = (float)H[i];
  v = v * scale[p * C_DIM + c] + shift[p * C_DIM + c];
  H[i] = (_Float16)fmaxf(v, 0.f);
}

__global__ __launch_bounds__(256)
void pool_kernel(const _Float16* __restrict__ H, _Float16* __restrict__ ctx) {
  int i = blockIdx.x * 256 + threadIdx.x;          // p*256*1024 + b*1024 + c
  if (i >= NPATCH * B_DIM * C_DIM) return;
  int c = i & (C_DIM - 1);
  int t = i >> 10;
  int b = t & (B_DIM - 1);
  int p = t >> 8;
  const _Float16* base = H + ((size_t)p * MROWS + (size_t)b * 9) * C_DIM + c;
  float s = 0.f;
  #pragma unroll
  for (int r = 0; r < 9; ++r) s += (float)base[(size_t)r * C_DIM];
  ctx[i] = (_Float16)(s * (1.f / 9.f));
}

// ---------------------------------------------------------------------------
extern "C" void kernel_launch(void* const* d_in, const int* in_sizes, int n_in,
                              void* d_out, int out_size, void* d_ws, size_t ws_size,
                              hipStream_t stream) {
  const float* x        = (const float*)d_in[0];
  const float* bn_gamma = (const float*)d_in[1];
  const float* bn_beta  = (const float*)d_in[2];
  const float* conv_w   = (const float*)d_in[3];
  const float* conv_b   = (const float*)d_in[4];
  const float* lin_w    = (const float*)d_in[5];
  const float* lin_b    = (const float*)d_in[6];
  float* out = (float*)d_out;

  char* ws = (char*)d_ws;
  const size_t SA = (size_t)NPATCH * MROWS * C_DIM * sizeof(_Float16); // 117,964,800 B
  _Float16* Ha    = (_Float16*)(ws);
  _Float16* Hb    = (_Float16*)(ws + SA);
  _Float16* Wc16  = (_Float16*)(ws + 2 * SA);
  _Float16* Wl16  = (_Float16*)(ws + 2 * SA + 3ull  * C_DIM * C_DIM * 2);
  _Float16* ctx16 = (_Float16*)(ws + 2 * SA + 15ull * C_DIM * C_DIM * 2);
  float* scale = (float*)(ws + 2 * SA + 15ull * C_DIM * C_DIM * 2
                             + (size_t)NPATCH * B_DIM * C_DIM * 2);
  float* shift = scale + NPATCH * C_DIM;
  float* psum  = shift + NPATCH * C_DIM;
  float* psq   = psum  + NPATCH * C_DIM;

  // fp16 weight copies
  { int n = 3  * C_DIM * C_DIM; f32_to_f16_kernel<<<(n + 255) / 256, 256, 0, stream>>>(conv_w, Wc16, n); }
  { int n = 12 * C_DIM * C_DIM; f32_to_f16_kernel<<<(n + 255) / 256, 256, 0, stream>>>(lin_w,  Wl16, n); }

  // gather 25 3x3 patches -> H0 [25][2304][1024] fp16
  { long long n = (long long)NPATCH * MROWS * C_DIM;
    pack_patches_kernel<<<(unsigned)((n + 255) / 256), 256, 0, stream>>>(x, Ha); }

  _Float16* Hc = Ha; _Float16* Hn = Hb;
  for (int l = 0; l < 3; ++l) {
    { int n = NPATCH * C_DIM * 2;   // psum & psq contiguous
      zero_f32_kernel<<<(n + 255) / 256, 256, 0, stream>>>(psum, n); }
    bn_partial_kernel<<<dim3(NPATCH, MROWS / 128), 256, 0, stream>>>(Hc, psum, psq);
    { int n = NPATCH * C_DIM;
      bn_final_kernel<<<(n + 255) / 256, 256, 0, stream>>>(psum, psq,
          bn_gamma + l * C_DIM, bn_beta + l * C_DIM, scale, shift); }
    { long long n = (long long)NPATCH * MROWS * C_DIM;
      bn_apply_kernel<<<(unsigned)((n + 255) / 256), 256, 0, stream>>>(Hc, scale, shift); }
    gemm_conv_kernel<<<dim3(C_DIM / 128, MROWS / 128, NPATCH), 256, 0, stream>>>(
        Hc, Wc16 + (size_t)l * C_DIM * C_DIM, conv_b + l * C_DIM, Hn, MROWS);
    _Float16* t = Hc; Hc = Hn; Hn = t;
  }

  { int n = NPATCH * B_DIM * C_DIM;
    pool_kernel<<<(n + 255) / 256, 256, 0, stream>>>(Hc, ctx16); }

  // replicate the reference's exact prediction enumeration order
  PredTab tab; int q = 0;
  for (int y1 = 0; y1 < 5; ++y1)
    for (int x1 = 0; x1 < 5; ++x1) {
      int y2 = y1 + 2, x2 = x1 + 2, p = y1 * 5 + x1;
      if (y2 == 2 || y2 == 3) for (int s = 0; s < 3; ++s) { if (y2 + s + 1 > 6) break; tab.e[q++] = (unsigned short)((p << 8) | (0 * 3 + s)); }
      if (y1 == 3 || y1 == 4) for (int s = 0; s < 3; ++s) { if (y1 - (s + 1) < 0) break; tab.e[q++] = (unsigned short)((p << 8) | (1 * 3 + s)); }
      if (x2 == 2 || x2 == 3) for (int s = 0; s < 3; ++s) { if (x2 + s + 1 > 6) break; tab.e[q++] = (unsigned short)((p << 8) | (2 * 3 + s)); }
      if (x1 == 3 || x1 == 4) for (int s = 0; s < 3; ++s) { if (x1 - (s + 1) < 0) break; tab.e[q++] = (unsigned short)((p << 8) | (3 * 3 + s)); }
    }
  gemm_head_kernel<<<dim3(C_DIM / 128, B_DIM / 128, 120), 256, 0, stream>>>(
      ctx16, Wl16, lin_b, out, tab);

  (void)in_sizes; (void)n_in; (void)out_size; (void)ws_size;
}
